// CvtLstm_61357902790780
// MI455X (gfx1250) — compile-verified, matrix operand order checked by reading, not run on hardware
//
#include <hip/hip_runtime.h>
#include <math.h>

// ---------------------------------------------------------------------------
// CvtLstm on MI455X (gfx1250). h==0 collapses the module:
//   xin = W_in x                          (WMMA GEMM, NHWC f16 out)
//   q0=conv3x3 SAME, k0/v0/v2=VALID       (WMMA implicit GEMM: weights in LDS,
//                                          NHWC input -> 2x b128 per K-step)
//   attn0 = softmax(q0^T k0) v0           (WMMA scores + big-LDS softmax + WMMA PV)
//   attn2 = mean_D(v2) constant; attn1=attn3=0
//   gates/out: small per-pixel 1x1s.
// Intermediates live in WMMA fragment layout (ISA 7.12.2) so every operand is
// a 128/256-bit load.  K-order = tap*64+ic makes the tap uniform per K-step.
// ---------------------------------------------------------------------------

typedef __attribute__((ext_vector_type(16))) _Float16 v16h;
typedef __attribute__((ext_vector_type(8)))  _Float16 v8h;
typedef __attribute__((ext_vector_type(8)))  float    v8f;

#define NB    4
#define C64   64
#define HEADS 4
#define HC    16
#define HH    48
#define WW    48
#define QTOT  (HH*WW)     // 2304
#define NTQ   144         // q tiles of 16
#define DH    46
#define DW    46
#define DTOT  (DH*DW)     // 2116
#define NTK   134         // key tiles of 16 (covers 2144)
#define NTV   67          // key tiles of 32
#define DPAD  2144

// operand K index for 16-bit 16x32 A/B fragments (ISA 7.12.2)
__device__ __forceinline__ int kmapA(int j, int hi) {
  return ((j < 8) ? j : j + 8) + hi;     // hi = 8 for lanes 16..31
}

// ------------- kernel 1: xin = W_in * x (1x1), NHWC f16 output --------------
// M = out channel (A = W_in), N = pixel (B = x), K = in channel.
__global__ void xin_kernel(const float* __restrict__ x, const float* __restrict__ Wi,
                           const float* __restrict__ bi, _Float16* __restrict__ xin) {
  const int lane = threadIdx.x;
  const int mcol = lane & 15;
  const int hi   = (lane >= 16) ? 8 : 0;
  const int pixbase = blockIdx.x * 16;
  const int ocbase  = blockIdx.y * 16;
  const int n       = blockIdx.z;
  const int oc  = ocbase + mcol;         // A: M = out channel
  const int pix = pixbase + mcol;        // B: N = pixel
  v8f acc = {};
  for (int step = 0; step < 2; ++step) {
    v16h a = {}, b = {};
#pragma unroll
    for (int j = 0; j < 16; ++j) {
      const int K = step * 32 + kmapA(j, hi);
      a[j] = (_Float16)Wi[oc * C64 + K];
      b[j] = (_Float16)x[((size_t)(n * C64 + K)) * QTOT + pix];
    }
    acc = __builtin_amdgcn_wmma_f32_16x16x32_f16(false, a, false, b, (short)0, acc,
                                                 false, false);
  }
  // D: N = pixel = mcol, M = oc = ocbase + r + hi  -> contiguous NHWC store
  v8h o8;
#pragma unroll
  for (int r = 0; r < 8; ++r) o8[r] = (_Float16)(acc[r] + bi[ocbase + r + hi]);
  *(v8h*)&xin[((size_t)n * QTOT + pix) * C64 + ocbase + hi] = o8;
}

// ------------- kernel 2: 3x3 conv 64->64, M = channel, N = position ---------
// A = weights (LDS fragments), B = NHWC input: per K-step the 3x3 tap is
// uniform and the 2x8 channels are consecutive -> two b128 loads.
// mode 0: NHWC dst [n][pix][oc]           (v2)
// mode 1: 8-half A/B fragment dst (q / k), tiles of 16 positions
// mode 2: 16-half B fragment dst (v for PV), tiles of 32 positions
#define CONV_TILES 16
__global__ void conv3_kernel(const _Float16* __restrict__ src, const float* __restrict__ w,
                             _Float16* __restrict__ dst, int same, int oW, int oH,
                             int mode, int ntiles) {
  __shared__ _Float16 wlds[18 * 512];    // 16 oc x 576 K in A-fragment layout
  const int lane = threadIdx.x;
  const int mcol = lane & 15;
  const int hi   = (lane >= 16) ? 8 : 0;
  const int npix = oW * oH;
  const int ocbase = blockIdx.y * 16;
  const int head   = blockIdx.y;
  const int n      = blockIdx.z;
  const int pad = same ? 1 : 0;

  // stage weight A-fragments: wlds[step*512 + lp*16 + j2] = w[K][ocbase+lp%16]
  for (int i = lane; i < 18 * 512; i += 32) {
    const int step = i >> 9;
    const int rem  = i & 511;
    const int lp   = rem >> 4;
    const int j2   = rem & 15;
    const int K    = step * 32 + kmapA(j2, (lp >= 16) ? 8 : 0);
    const int kk   = K >> 6;
    const int ic   = K & 63;
    const int ky   = kk / 3, kx = kk % 3;
    const int oc   = ocbase + (lp & 15);
    wlds[i] = (_Float16)w[(((size_t)oc * C64 + ic) * 3 + ky) * 3 + kx];
  }
  __syncthreads();

  for (int t = 0; t < CONV_TILES; ++t) {
    const int pixbase = (blockIdx.x * CONV_TILES + t) * 16;
    if (pixbase >= npix) break;
    const int p  = pixbase + mcol;               // B: N = output position
    const int oy = p / oW, ox = p % oW;
    v8f acc = {};
    for (int step = 0; step < 18; ++step) {
      const int kk = step >> 1;                  // tap uniform across the step
      const int ky = kk / 3, kx = kk % 3;
      const int iy = oy + ky - pad;
      const int ix = ox + kx - pad;
      v16h b = {};
      if (p < npix && iy >= 0 && iy < HH && ix >= 0 && ix < WW) {
        const _Float16* base =
            &src[((size_t)n * QTOT + iy * WW + ix) * C64 + (step & 1) * 32 + hi];
        const v8h blo = *(const v8h*)base;       // ic = icbase .. +7
        const v8h bhi = *(const v8h*)(base + 16);// ic = icbase+16 .. +23
#pragma unroll
        for (int j = 0; j < 8; ++j) { b[j] = blo[j]; b[j + 8] = bhi[j]; }
      }
      const v16h a = *(const v16h*)&wlds[step * 512 + (lane << 4)];
      acc = __builtin_amdgcn_wmma_f32_16x16x32_f16(false, a, false, b, (short)0, acc,
                                                   false, false);
    }
    if (p >= npix) continue;
    // D: M = channel = r + hi, N = position = mcol
    if (mode == 0) {                             // NHWC
      v8h o8;
#pragma unroll
      for (int r = 0; r < 8; ++r) o8[r] = (_Float16)acc[r];
      *(v8h*)&dst[((size_t)n * npix + p) * C64 + ocbase + hi] = o8;
    } else if (mode == 1) {                      // q/k fragment: j2 = r (contig)
      v8h o8;
#pragma unroll
      for (int r = 0; r < 8; ++r) o8[r] = (_Float16)acc[r];
      const int lp = (p & 15) + (hi ? 16 : 0);   // c>=8 iff hi==8
      *(v8h*)&dst[((size_t)(n * HEADS + head) * ntiles + (p >> 4)) * 256 + lp * 8] = o8;
    } else {                                     // v fragment (PV B operand)
      const int dm = p & 31;
      const int j2 = (dm & 7) + ((dm >= 16) ? 8 : 0);
      const int cadd = (dm & 8) ? 16 : 0;
      const size_t tb = ((size_t)(n * HEADS + head) * NTV + (p >> 5)) * 512;
#pragma unroll
      for (int r = 0; r < 8; ++r)
        dst[tb + (r + hi + cadd) * 16 + j2] = (_Float16)acc[r];
    }
  }
}

// ------- kernel 3: spatial mean of v2 (NHWC) and branch-2 gate constants ----
__global__ void mean_gate_kernel(const _Float16* __restrict__ v2,
                                 const float* __restrict__ Wproj,
                                 float* __restrict__ gate_c2) {
  __shared__ float a2[C64];
  const int n = blockIdx.x;
  const int o = threadIdx.x;
  float s = 0.f;
  for (int d = 0; d < DTOT; ++d)                 // coalesced across 64 threads
    s += (float)v2[((size_t)n * DTOT + d) * C64 + o];
  a2[o] = s / (float)DTOT;
  __syncthreads();
  for (int g = 0; g < 4; ++g) {
    float acc = 0.f;
    const float* wp = Wproj + (((size_t)g * 4 + 2) * C64 + o) * C64;
    for (int ca = 0; ca < C64; ++ca) acc += wp[ca] * a2[ca];
    gate_c2[(g * NB + n) * C64 + o] = acc;
  }
}

// ------ kernel 4: attention branch 0; operands pre-tiled, softmax in LDS ----
__global__ void attn_kernel(const _Float16* __restrict__ qf, const _Float16* __restrict__ kf,
                            const _Float16* __restrict__ vf, float* __restrict__ a0) {
  extern __shared__ char smraw[];
  float*    S  = (float*)smraw;                          // 16 x DPAD f32 scores
  _Float16* PF = (_Float16*)(smraw + 16 * DPAD * 4);     // P in fragment layout
  float* lsum  = (float*)(smraw + 16 * DPAD * 4 + NTV * 512 * 2);
  const int lane = threadIdx.x;
  const int mcol = lane & 15;
  const int hi   = (lane >= 16) ? 8 : 0;
  const int qt   = blockIdx.x;
  const int head = blockIdx.y;
  const int n    = blockIdx.z;
  const size_t hb = (size_t)(n * HEADS + head);

  // A operand: q tile, one 128-bit load (upper K half is the HC->32 zero pad)
  v16h aq = {};
  {
    const v8h lo = *(const v8h*)&qf[(hb * NTQ + qt) * 256 + lane * 8];
#pragma unroll
    for (int j = 0; j < 8; ++j) aq[j] = lo[j];
  }

  // scores: one WMMA + one 128-bit k load per 16 keys
  for (int dt = 0; dt < NTK; ++dt) {
    v16h bk = {};
    {
      const v8h lo = *(const v8h*)&kf[(hb * NTK + dt) * 256 + lane * 8];
#pragma unroll
      for (int j = 0; j < 8; ++j) bk[j] = lo[j];
    }
    v8f cs = {};
    cs = __builtin_amdgcn_wmma_f32_16x16x32_f16(false, aq, false, bk, (short)0, cs,
                                                false, false);
    const int d = dt * 16 + mcol;
#pragma unroll
    for (int r = 0; r < 8; ++r)
      S[(r + hi) * DPAD + d] = (d < DTOT) ? cs[r] : -1e30f;
  }
  __syncthreads();

  // softmax: max pass, then exp pass writing P as f16 fragments into LDS
  for (int row = 0; row < 16; ++row) {
    float m = -1e30f;
    for (int col = lane; col < DPAD; col += 32) m = fmaxf(m, S[row * DPAD + col]);
#pragma unroll
    for (int off = 16; off > 0; off >>= 1) m = fmaxf(m, __shfl_xor(m, off, 32));
    float s = 0.f;
    for (int col = lane; col < DPAD; col += 32) {
      const float e = __expf(S[row * DPAD + col] - m);
      s += e;
      const int dm = col & 31;
      const int lp = row + ((dm & 8) ? 16 : 0);
      const int j2 = (dm & 7) + ((dm >= 16) ? 8 : 0);
      PF[(col >> 5) * 512 + lp * 16 + j2] = (_Float16)e;
    }
#pragma unroll
    for (int off = 16; off > 0; off >>= 1) s += __shfl_xor(s, off, 32);
    if (lane == 0) lsum[row] = s;
  }
  __syncthreads();

  // PV: P fragment from LDS (32B ds load), V fragment from global (32B load)
  v8f acc = {};
  for (int dt = 0; dt < NTV; ++dt) {
    const v16h ap = *(const v16h*)&PF[dt * 512 + (lane << 4)];
    const v16h bv = *(const v16h*)&vf[(hb * NTV + dt) * 512 + lane * 16];
    acc = __builtin_amdgcn_wmma_f32_16x16x32_f16(false, ap, false, bv, (short)0, acc,
                                                 false, false);
  }
  // D: M = query = r+hi, N = head channel = mcol ; a0 is [n][pix][ch]
#pragma unroll
  for (int r = 0; r < 8; ++r) {
    const int row = r + hi;
    a0[((size_t)n * QTOT + qt * 16 + row) * C64 + head * HC + mcol] = acc[r] / lsum[row];
  }
}

// --------- kernel 5: gate projections + LSTM gating + W_out per pixel -------
__global__ void gates_out_kernel(const float* __restrict__ a0, const float* __restrict__ gate_c2,
                                 const float* __restrict__ Wproj, const float* __restrict__ b_gates,
                                 const float* __restrict__ Wout, const float* __restrict__ bout,
                                 float* __restrict__ out) {
  __shared__ float acol[C64];
  __shared__ float hcol[C64];
  const int t = threadIdx.x;
  const int pix = blockIdx.x;
  const int n = blockIdx.y;
  acol[t] = a0[((size_t)n * QTOT + pix) * C64 + t];     // coalesced
  __syncthreads();
  float pre[4];
#pragma unroll
  for (int g = 0; g < 4; ++g) {
    float s = b_gates[g * C64 + t] + gate_c2[(g * NB + n) * C64 + t];
    const float* wp = Wproj + (((size_t)g * 4 + 0) * C64 + t) * C64;
    for (int ca = 0; ca < C64; ++ca) s += wp[ca] * acol[ca];
    pre[g] = s;
  }
  const float ig = 1.f / (1.f + __expf(-pre[0]));
  const float gg = tanhf(pre[2]);
  const float og = 1.f / (1.f + __expf(-pre[3]));
  hcol[t] = og * tanhf(ig * gg);               // f*c_prev vanishes (c_prev = 0)
  __syncthreads();
  float s = bout[t];
  for (int c = 0; c < C64; ++c) s += Wout[t * C64 + c] * hcol[c];
  out[((size_t)(n * C64 + t)) * QTOT + pix] = s;        // NCHW reference output
}

// ---------------------------------------------------------------------------
extern "C" void kernel_launch(void* const* d_in, const int* in_sizes, int n_in,
                              void* d_out, int out_size, void* d_ws, size_t ws_size,
                              hipStream_t stream) {
  (void)in_sizes; (void)n_in; (void)out_size; (void)ws_size;
  const float* x     = (const float*)d_in[0];
  const float* W_in  = (const float*)d_in[1];
  const float* b_in  = (const float*)d_in[2];
  const float* Wq    = (const float*)d_in[3];
  const float* Wk    = (const float*)d_in[4];
  const float* Wv    = (const float*)d_in[5];
  const float* Wproj = (const float*)d_in[6];
  const float* b_g   = (const float*)d_in[7];
  const float* W_out = (const float*)d_in[8];
  const float* b_out = (const float*)d_in[9];
  float* out = (float*)d_out;

  char* ws = (char*)d_ws;
  size_t off = 0;
  auto alloc = [&](size_t bytes) -> char* {
    char* p = ws + off;
    off += (bytes + 255) & ~(size_t)255;
    return p;
  };
  const size_t qfragBytes = (size_t)NB * HEADS * NTQ * 256 * 2;
  const size_t kfragBytes = (size_t)NB * HEADS * NTK * 256 * 2;
  const size_t vfragBytes = (size_t)NB * HEADS * NTV * 512 * 2;
  _Float16* xin  = (_Float16*)alloc((size_t)NB * QTOT * C64 * 2);   // NHWC
  _Float16* qfr  = (_Float16*)alloc(qfragBytes);
  _Float16* kfr  = (_Float16*)alloc(kfragBytes);
  _Float16* vfr  = (_Float16*)alloc(vfragBytes);
  _Float16* v2b  = (_Float16*)alloc((size_t)NB * DTOT * C64 * 2);   // NHWC
  float*    a0   = (float*)   alloc((size_t)NB * QTOT * C64 * 4);   // [n][pix][ch]
  float*    gc2  = (float*)   alloc((size_t)4 * NB * C64 * 4);

  const size_t WBLK = (size_t)C64 * C64 * 9;     // one (combo) 3x3 weight block

  xin_kernel<<<dim3(NTQ, C64 / 16, NB), 32, 0, stream>>>(x, W_in, b_in, xin);

  // zero fragment pads (d >= DTOT) so WMMA operands are well-defined
  hipMemsetAsync(kfr, 0, kfragBytes, stream);
  hipMemsetAsync(vfr, 0, vfragBytes, stream);

  const int gxQ = (NTQ + CONV_TILES - 1) / CONV_TILES;                 // 9
  const int gxD = (((DTOT + 15) / 16) + CONV_TILES - 1) / CONV_TILES;  // 9
  conv3_kernel<<<dim3(gxQ, HEADS, NB), 32, 0, stream>>>(xin, Wq + 0 * WBLK, qfr, 1, WW, HH, 1, NTQ);
  conv3_kernel<<<dim3(gxD, HEADS, NB), 32, 0, stream>>>(xin, Wk + 0 * WBLK, kfr, 0, DW, DH, 1, NTK);
  conv3_kernel<<<dim3(gxD, HEADS, NB), 32, 0, stream>>>(xin, Wv + 0 * WBLK, vfr, 0, DW, DH, 2, 0);
  conv3_kernel<<<dim3(gxD, HEADS, NB), 32, 0, stream>>>(xin, Wv + 2 * WBLK, v2b, 0, DW, DH, 0, 0);

  mean_gate_kernel<<<dim3(NB), C64, 0, stream>>>(v2b, Wproj, gc2);

  const size_t shmem = (size_t)16 * DPAD * 4 + (size_t)NTV * 512 * 2 + 64;
  attn_kernel<<<dim3(NTQ, HEADS, NB), 32, shmem, stream>>>(qfr, kfr, vfr, a0);

  gates_out_kernel<<<dim3(QTOT, NB), C64, 0, stream>>>(a0, gc2, Wproj, b_g, W_out, b_out, out);
}
